// OmniAID_LoRA_33337536151853
// MI455X (gfx1250) — matrix operand order repaired and checked by reading, not compiled, not run on hardware
//
#include <hip/hip_runtime.h>

// ---------------------------------------------------------------------------
// OmniAID LoRA-MoE fused forward for gfx1250 (MI455X, wave32, WMMA bf16)
//   out[b,s,o] = bias[o] + sum_d x*W^T + sum_k gate * ((x A_ek^T) B_ek^T)
// Shapes: B=64 S=577 D=O=1024 E=8 R=16 K=2
// ---------------------------------------------------------------------------

typedef __attribute__((ext_vector_type(16))) __bf16 v16bf;
typedef __attribute__((ext_vector_type(8)))  float  v8f;
typedef int v4i_vs __attribute__((vector_size(16)));   // matches builtin V4i

#define BN 64
#define SN 577
#define DN 1024
#define EN 8
#define RN 16
#define KT 2

#if defined(__has_builtin)
#  if __has_builtin(__builtin_amdgcn_global_load_async_to_lds_b128)
#    define HAVE_ASYNC_LDS 1
#  endif
#  if __has_builtin(__builtin_amdgcn_s_wait_asynccnt)
#    define HAVE_WAIT_ASYNC_BUILTIN 1
#  endif
#endif
#ifndef HAVE_ASYNC_LDS
#  define HAVE_ASYNC_LDS 0
#endif
#ifndef HAVE_WAIT_ASYNC_BUILTIN
#  define HAVE_WAIT_ASYNC_BUILTIN 0
#endif

union FragB16 { v16bf v; uint4 q[2]; };

__device__ __forceinline__ unsigned short f2bf(float f) {
    union { float f; unsigned u; } v; v.f = f;
    unsigned u = v.u;
    u += 0x7fffu + ((u >> 16) & 1u);   // round-to-nearest-even
    return (unsigned short)(u >> 16);
}

__device__ __forceinline__ void wait_async_zero() {
#if HAVE_ASYNC_LDS
#  if HAVE_WAIT_ASYNC_BUILTIN
    __builtin_amdgcn_s_wait_asynccnt(0);
#  else
    asm volatile("s_wait_asynccnt 0x0" ::: "memory");
#  endif
#endif
}

// ---------------- Kernel 0: f32 -> bf16 conversion (vectorized x4) ---------
__global__ void cvt_bf16_kernel(const float* __restrict__ src,
                                unsigned short* __restrict__ dst, int n4) {
    int i = blockIdx.x * blockDim.x + threadIdx.x;
    if (i < n4) {
        float4 f = ((const float4*)src)[i];
        ushort4 o;
        o.x = f2bf(f.x); o.y = f2bf(f.y); o.z = f2bf(f.z); o.w = f2bf(f.w);
        ((ushort4*)dst)[i] = o;
    }
}

// ---------------- Kernel 1: xa[b,s,kk*16+r] = gate * (x @ A_e^T), bf16 -----
// one wave per (m-tile, b, kk); WMMA over K=D in steps of 32
__global__ __launch_bounds__(32) void lora_xa_kernel(
    const unsigned short* __restrict__ xb,   // [B,S,D] bf16
    const unsigned short* __restrict__ ab,   // [E,R,D] bf16
    const long long*      __restrict__ tki,  // [B,K]
    const float*          __restrict__ gates,// [B,K]
    unsigned short*       __restrict__ xa)   // [B,S,32] bf16
{
    const int lane   = threadIdx.x;
    const int lane16 = lane & 15;
    const int hi     = lane >> 4;           // 0 or 1
    const int b      = blockIdx.y;
    const int kk     = blockIdx.z;
    const int s0     = blockIdx.x * 16;
    const int e      = (int)tki[b * KT + kk];
    const float g    = gates[b * KT + kk];

    const int srow = min(s0 + lane16, SN - 1);
    const unsigned short* xrow = xb + ((size_t)b * SN + srow) * DN;
    const unsigned short* arow = ab + ((size_t)e * RN + lane16) * DN;
    const int abase = hi ? 8  : 0;          // A-frag K chunk base
    const int bbase = hi ? 16 : 0;          // unused; B frag (w) also uses abase

    v8f acc = {0.f,0.f,0.f,0.f,0.f,0.f,0.f,0.f};
    for (int d0 = 0; d0 < DN; d0 += 32) {
        FragB16 a, w;
        a.q[0] = *(const uint4*)(xrow + d0 + abase);
        a.q[1] = *(const uint4*)(xrow + d0 + abase + 16);
        w.q[0] = *(const uint4*)(arow + d0 + abase);
        w.q[1] = *(const uint4*)(arow + d0 + abase + 16);
        acc = __builtin_amdgcn_wmma_f32_16x16x32_bf16(
            false, a.v, false, w.v, (short)0, acc, false, false);
    }
    #pragma unroll
    for (int i = 0; i < 8; ++i) {
        int s = s0 + i + hi * 8;
        if (s < SN)
            xa[((size_t)b * SN + s) * (KT * RN) + kk * RN + lane16] =
                f2bf(g * acc[i]);
    }
}

// ---------------- Kernel 2: main GEMM + fused LoRA epilogue + bias ---------
// block = 256 threads (8 waves); wave w owns 32 rows; block tile 256 x 64.
// W tile (64 cols x 64 K) staged in LDS via async global->LDS when available,
// shared by all 8 waves; inner 2 K-steps of 32 between barriers.
__global__ __launch_bounds__(256) void moe_gemm_kernel(
    const unsigned short* __restrict__ xb,   // [B,S,D] bf16
    const unsigned short* __restrict__ wb,   // [O,D]   bf16
    const unsigned short* __restrict__ bb,   // [E,O,R] bf16
    const unsigned short* __restrict__ xa,   // [B,S,32] bf16 (gate-scaled)
    const long long*      __restrict__ tki,  // [B,K]
    const float*          __restrict__ bias, // [O]
    float*                __restrict__ out)  // [B,S,O]
{
    __shared__ uint4 ldsW[512];              // 64 cols x 64 K bf16 = 8 KB

    const int tid    = threadIdx.x;
    const int lane   = tid & 31;
    const int lane16 = lane & 15;
    const int hi     = lane >> 4;
    const int wave   = tid >> 5;
    const int o0     = blockIdx.x * 64;
    const int s0     = blockIdx.y * 256 + wave * 32;
    const int b      = blockIdx.z;

    const int srow0 = min(s0 +      lane16, SN - 1);
    const int srow1 = min(s0 + 16 + lane16, SN - 1);
    const unsigned short* xrow0 = xb + ((size_t)b * SN + srow0) * DN;
    const unsigned short* xrow1 = xb + ((size_t)b * SN + srow1) * DN;
    const int abase = hi ? 16 : 0;

    v8f acc[2][4];
    #pragma unroll
    for (int mg = 0; mg < 2; ++mg)
        #pragma unroll
        for (int nn = 0; nn < 4; ++nn)
            acc[mg][nn] = (v8f){0.f,0.f,0.f,0.f,0.f,0.f,0.f,0.f};

    const int cW  = tid >> 2;                // 0..63: W tile column
    const int chW = tid & 3;                 // 0..3 : 32B chunk within row

    for (int d0 = 0; d0 < DN; d0 += 64) {
        // ---- stage W tile: lds_w[c][k] = wb[o0+c][d0+k], k in [0,64)
        {
            const unsigned short* wrow =
                wb + (size_t)(o0 + cW) * DN + d0 + chW * 16;
            const int li = (cW << 3) + (chW << 1);
#if HAVE_ASYNC_LDS
            __builtin_amdgcn_global_load_async_to_lds_b128(
                (__attribute__((address_space(1))) v4i_vs*)wrow,
                (__attribute__((address_space(3))) v4i_vs*)&ldsW[li], 0, 0);
            __builtin_amdgcn_global_load_async_to_lds_b128(
                (__attribute__((address_space(1))) v4i_vs*)(wrow + 8),
                (__attribute__((address_space(3))) v4i_vs*)&ldsW[li + 1], 0, 0);
            wait_async_zero();
#else
            ldsW[li]     = *(const uint4*)(wrow);
            ldsW[li + 1] = *(const uint4*)(wrow + 8);
#endif
        }
        __syncthreads();

        if (d0 + 64 < DN) {
            __builtin_prefetch(xrow0 + d0 + 64, 0, 0); // global_prefetch_b8
            __builtin_prefetch(xrow1 + d0 + 64, 0, 0);
        }

        #pragma unroll
        for (int kh = 0; kh < 2; ++kh) {
            const int ks  = kh * 32;
            const int ks4 = kh * 4;
            FragB16 a0, a1;
            a0.q[0] = *(const uint4*)(xrow0 + d0 + ks + abase);
            a0.q[1] = *(const uint4*)(xrow0 + d0 + ks + abase + 8);
            a1.q[0] = *(const uint4*)(xrow1 + d0 + ks + abase);
            a1.q[1] = *(const uint4*)(xrow1 + d0 + ks + abase + 8);

            #pragma unroll
            for (int nn = 0; nn < 4; ++nn) {
                FragB16 w;
                const int base = ((nn * 16 + lane16) << 3) + ks4 + hi * 2;
                w.q[0] = ldsW[base];         // ds_load_b128
                w.q[1] = ldsW[base + 1];
                acc[0][nn] = __builtin_amdgcn_wmma_f32_16x16x32_bf16(
                    false, a0.v, false, w.v, (short)0, acc[0][nn], false, false);
                acc[1][nn] = __builtin_amdgcn_wmma_f32_16x16x32_bf16(
                    false, a1.v, false, w.v, (short)0, acc[1][nn], false, false);
            }
        }
        __syncthreads();
    }

    // ---- fused LoRA epilogue: one WMMA per 16x16 tile (K = 2 experts x R=16)
    const int e0 = (int)tki[b * KT + 0];
    const int e1 = (int)tki[b * KT + 1];
    const int eh = hi ? e1 : e0;             // B-frag lane holds 16 contig K

    FragB16 a2[2];
    {
        const unsigned short* xar0 = xa + ((size_t)b * SN + srow0) * (KT * RN);
        const unsigned short* xar1 = xa + ((size_t)b * SN + srow1) * (KT * RN);
        a2[0].q[0] = *(const uint4*)(xar0 + abase);
        a2[0].q[1] = *(const uint4*)(xar0 + abase + 8);
        a2[1].q[0] = *(const uint4*)(xar1 + abase);
        a2[1].q[1] = *(const uint4*)(xar1 + abase + 8);
    }

    #pragma unroll
    for (int nn = 0; nn < 4; ++nn) {
        const int o = o0 + nn * 16 + lane16;
        FragB16 b2;
        const unsigned short* brow = bb + ((size_t)eh * DN + o) * RN;
        b2.q[0] = *(const uint4*)(brow);
        b2.q[1] = *(const uint4*)(brow + 8);
        const float bv = bias[o];

        #pragma unroll
        for (int mg = 0; mg < 2; ++mg) {
            acc[mg][nn] = __builtin_amdgcn_wmma_f32_16x16x32_bf16(
                false, a2[mg].v, false, b2.v, (short)0, acc[mg][nn], false, false);
            #pragma unroll
            for (int i = 0; i < 8; ++i) {
                const int s = s0 + mg * 16 + i + hi * 8;
                if (s < SN)
                    out[((size_t)b * SN + s) * DN + o] = acc[mg][nn][i] + bv;
            }
        }
    }
}

// ---------------------------------------------------------------------------
extern "C" void kernel_launch(void* const* d_in, const int* in_sizes, int n_in,
                              void* d_out, int out_size, void* d_ws, size_t ws_size,
                              hipStream_t stream) {
    (void)in_sizes; (void)n_in; (void)out_size; (void)ws_size;
    const float*     x     = (const float*)d_in[0];
    const long long* tki   = (const long long*)d_in[1];
    const float*     gates = (const float*)d_in[2];
    const float*     wf    = (const float*)d_in[3];
    const float*     A_all = (const float*)d_in[4];
    const float*     B_all = (const float*)d_in[5];
    const float*     bias  = (const float*)d_in[6];
    float* out = (float*)d_out;

    const size_t XB_E = (size_t)BN * SN * DN;   // 37,814,272
    const size_t WB_E = (size_t)DN * DN;        //  1,048,576
    const size_t AB_E = (size_t)EN * RN * DN;   //    131,072
    const size_t BB_E = (size_t)EN * DN * RN;   //    131,072

    unsigned short* xb = (unsigned short*)d_ws;
    unsigned short* wb = xb + XB_E;
    unsigned short* ab = wb + WB_E;
    unsigned short* bb = ab + AB_E;
    unsigned short* xa = bb + BB_E;             // [B,S,32] bf16

    // 0) f32 -> bf16 conversions
    cvt_bf16_kernel<<<(int)(XB_E/4 + 255)/256, 256, 0, stream>>>(x,     xb, (int)(XB_E/4));
    cvt_bf16_kernel<<<(int)(WB_E/4 + 255)/256, 256, 0, stream>>>(wf,    wb, (int)(WB_E/4));
    cvt_bf16_kernel<<<(int)(AB_E/4 + 255)/256, 256, 0, stream>>>(A_all, ab, (int)(AB_E/4));
    cvt_bf16_kernel<<<(int)(BB_E/4 + 255)/256, 256, 0, stream>>>(B_all, bb, (int)(BB_E/4));

    // 1) LoRA stage 1: xa = gate * (x @ A_e^T)   (37 m-tiles of 16 rows)
    lora_xa_kernel<<<dim3(37, BN, KT), 32, 0, stream>>>(xb, ab, tki, gates, xa);

    // 2) main GEMM + fused LoRA stage 2 + bias
    //    grid: 16 n-blocks(64) x 3 m-blocks(256) x 64 batches
    moe_gemm_kernel<<<dim3(16, 3, BN), 256, 0, stream>>>(xb, wb, bb, xa, tki, bias, out);
}
